// NAMTMAE_55138790146726
// MI455X (gfx1250) — matrix operand order repaired and checked by reading, not compiled, not run on hardware
//
#include <hip/hip_runtime.h>
#include <hip/hip_bf16.h>
#include <math.h>
#include <stdint.h>

typedef __bf16 bf16_t;
typedef _Float16 f16_t;
typedef __attribute__((ext_vector_type(16))) __bf16   v16bf;
typedef __attribute__((ext_vector_type(16))) _Float16 v16h;
typedef __attribute__((ext_vector_type(8)))  float    v8f;

constexpr int S_ = 256, B_ = 32, D_ = 512, T_ = 4, HD_ = 128, L_ = 32, V_ = 8192, H_ = 256, G3_ = 768;

union FragBF { v16bf v; uint4 q[2]; };
union FragH  { v16h  v; uint4 q[2]; };
union Acc8   { v8f v; float f[8]; };

__device__ __forceinline__ float sigmoidf_(float x) { return 1.f / (1.f + __expf(-x)); }

// ---- CDNA5 async global->LDS copy (ASYNCcnt-tracked), per CDNA5 ISA ----
__device__ __forceinline__ unsigned lds_off(const void* p) {
  return (unsigned)(uintptr_t)p;   // generic LDS addr: [31:0] is the LDS byte offset
}
__device__ __forceinline__ void async_load_b128(unsigned ldsaddr, const void* gaddr) {
  asm volatile("global_load_async_to_lds_b128 %0, %1, off"
               :: "v"(ldsaddr), "v"(gaddr) : "memory");
}
__device__ __forceinline__ void wait_async0() {
  asm volatile("s_wait_asynccnt 0x0" ::: "memory");
}

// ---------------------------------------------------------------------------
// C[M,N] = A[M,K](bf16) x W[N,K]^T(bf16) + bias, via v_wmma_f32_16x16x32_bf16.
// The block's 16xK A-tile is staged to LDS once via async-to-LDS and shared by
// all 8 waves (8x A-traffic reduction); W streams from global/L2 per wave.
// STORE: 0 = f32 row-major, 1 = bf16 row-major, 2 = logits transpose to [B,V,S] f32
// ---------------------------------------------------------------------------
template<int STORE>
__global__ __launch_bounds__(256)
void gemm_bf16(const bf16_t* __restrict__ A, const bf16_t* __restrict__ W,
               const float* __restrict__ bias, void* __restrict__ Cout,
               int M, int N, int K) {
  __shared__ bf16_t smA[16 * 512];            // K <= 512
  const int m0 = blockIdx.y * 16;
  // stage A tile: 16 rows x K, 16B chunks, all threads participate
  const int chunks = (16 * K) >> 3;           // 8 bf16 per chunk
  for (int i = threadIdx.x; i < chunks; i += 256) {
    const int e = i << 3;
    const int row = e / K, col = e % K;
    async_load_b128(lds_off(&smA[e]), A + (size_t)(m0 + row) * K + col);
  }
  wait_async0();
  __syncthreads();

  const int lane = threadIdx.x & 31;
  const int wv   = threadIdx.x >> 5;
  const int n0 = blockIdx.x * 128 + wv * 16;
  if (n0 < N) {                                // wave-uniform, no barriers below
    const int g = lane >> 4;                   // lane half
    const int c = lane & 15;
    const bf16_t* Arow = smA + (size_t)c * K;  // A frag row = lane&15 (from LDS)
    const bf16_t* Wrow = W + (size_t)(n0 + c) * K;
    Acc8 acc;
    const float bb = bias ? bias[n0 + c] : 0.f;
    #pragma unroll
    for (int j = 0; j < 8; ++j) acc.f[j] = bb;
    for (int kk = 0; kk < K; kk += 32) {
      FragBF af, wf;
      af.q[0] = *(const uint4*)(Arow + kk + g * 8);        // k = g*8 + 0..7
      af.q[1] = *(const uint4*)(Arow + kk + 16 + g * 8);   // k = 16 + g*8 + 0..7
      wf.q[0] = *(const uint4*)(Wrow + kk + g * 8);        // k = g*8 + 0..7
      wf.q[1] = *(const uint4*)(Wrow + kk + 16 + g * 8);   // k = 16 + g*8 + 0..7
      acc.v = __builtin_amdgcn_wmma_f32_16x16x32_bf16(false, af.v, false, wf.v,
                                                      (short)0, acc.v, false, false);
    }
    #pragma unroll
    for (int j = 0; j < 8; ++j) {
      const int row = m0 + g * 8 + j;          // C: lanes0-15 M=j, lanes16-31 M=8+j
      const int col = n0 + c;
      if (STORE == 0)      ((float*) Cout)[(size_t)row * N + col] = acc.f[j];
      else if (STORE == 1) ((bf16_t*)Cout)[(size_t)row * N + col] = (bf16_t)acc.f[j];
      else {
        const int s = row >> 5, b = row & 31;  // row = s*B + b
        ((float*)Cout)[((size_t)b * V_ + col) * S_ + s] = acc.f[j];
      }
    }
  }
}

// ---------------------------------------------------------------------------
// Small helper kernels
// ---------------------------------------------------------------------------
__global__ void cvt_bf16(const float* __restrict__ x, bf16_t* __restrict__ y, int n) {
  int i = blockIdx.x * 256 + threadIdx.x;
  if (i < n) y[i] = (bf16_t)x[i];
}
__global__ void cvt_f16(const float* __restrict__ x, f16_t* __restrict__ y, int n) {
  int i = blockIdx.x * 256 + threadIdx.x;
  if (i < n) y[i] = (f16_t)x[i];
}

// Wk^T: y[k*128+c] = x[c*128+k]
__global__ void cvt_transpose128(const float* __restrict__ x, bf16_t* __restrict__ y) {
  int i = blockIdx.x * 256 + threadIdx.x;
  if (i < HD_ * HD_) {
    int k = i >> 7, c = i & 127;
    y[i] = (bf16_t)x[c * HD_ + k];
  }
}

// Pad Wa [40,512] -> [48,512] bf16 (zero rows 40..47) and ba -> 48 f32
__global__ void pad_wa(const float* __restrict__ wa, const float* __restrict__ ba,
                       bf16_t* __restrict__ wa_h, float* __restrict__ ba_pad) {
  int i = blockIdx.x * 256 + threadIdx.x;
  if (i < 48 * D_) {
    int n = i / D_;
    wa_h[i] = (n < 40) ? (bf16_t)wa[(size_t)n * D_ + (i % D_)] : (bf16_t)0.f;
  }
  if (i < 48) ba_pad[i] = (i < 40) ? ba[i] : 0.f;
}

// Gather embedding rows -> bf16 [S*B, D]; row r = s*B + b; input is [B,S]
__global__ void gather_emb(const int* __restrict__ inp, const float* __restrict__ emb,
                           bf16_t* __restrict__ rows) {
  int r = blockIdx.x;
  int s = r >> 5, b = r & 31;
  int tok = inp[b * S_ + s];
  const float* e = emb + (size_t)tok * D_;
  bf16_t* o = rows + (size_t)r * D_;
  o[threadIdx.x]       = (bf16_t)e[threadIdx.x];
  o[threadIdx.x + 256] = (bf16_t)e[threadIdx.x + 256];
}

// LayerNorm (+optional relu) over rows of 512, f32 in -> bf16 out
__global__ __launch_bounds__(256)
void ln_kernel(const float* __restrict__ X, const float* __restrict__ gw,
               const float* __restrict__ bw, bf16_t* __restrict__ Y, int relu) {
  __shared__ float red[256];
  const int r = blockIdx.x, tid = threadIdx.x;
  const float* x = X + (size_t)r * D_;
  float a = x[tid], b = x[tid + 256];
  red[tid] = a + b; __syncthreads();
  for (int off = 128; off; off >>= 1) { if (tid < off) red[tid] += red[tid + off]; __syncthreads(); }
  float mean = red[0] * (1.f / 512.f);
  __syncthreads();
  float da = a - mean, db = b - mean;
  red[tid] = da * da + db * db; __syncthreads();
  for (int off = 128; off; off >>= 1) { if (tid < off) red[tid] += red[tid + off]; __syncthreads(); }
  float inv = rsqrtf(red[0] * (1.f / 512.f) + 1e-5f);
  float y0 = da * inv * gw[tid] + bw[tid];
  float y1 = db * inv * gw[tid + 256] + bw[tid + 256];
  if (relu) { y0 = fmaxf(y0, 0.f); y1 = fmaxf(y1, 0.f); }
  Y[(size_t)r * D_ + tid]       = (bf16_t)y0;
  Y[(size_t)r * D_ + tid + 256] = (bf16_t)y1;
}

// actions[r,48] -> dr(softmax4), dw(softmax4), rw(sigmoid2) at [r,40]
__global__ void act_proc(const float* __restrict__ act, float* __restrict__ outp) {
  int i = blockIdx.x * 256 + threadIdx.x;
  if (i >= (S_ * B_) * T_) return;
  int r = i >> 2, t = i & 3;
  const float* a = act + (size_t)r * 48 + t * 10;
  float* o = outp + (size_t)r * 40 + t * 10;
  {
    float m = fmaxf(fmaxf(a[0], a[1]), fmaxf(a[2], a[3]));
    float e0 = __expf(a[0] - m), e1 = __expf(a[1] - m), e2 = __expf(a[2] - m), e3 = __expf(a[3] - m);
    float inv = 1.f / (e0 + e1 + e2 + e3);
    o[0] = e0 * inv; o[1] = e1 * inv; o[2] = e2 * inv; o[3] = e3 * inv;
  }
  {
    float m = fmaxf(fmaxf(a[4], a[5]), fmaxf(a[6], a[7]));
    float e0 = __expf(a[4] - m), e1 = __expf(a[5] - m), e2 = __expf(a[6] - m), e3 = __expf(a[7] - m);
    float inv = 1.f / (e0 + e1 + e2 + e3);
    o[4] = e0 * inv; o[5] = e1 * inv; o[6] = e2 * inv; o[7] = e3 * inv;
  }
  o[8] = sigmoidf_(a[8]);
  o[9] = sigmoidf_(a[9]);
}

// ---------------------------------------------------------------------------
// Persistent bidirectional GRU scan. 4 blocks = (tm,dir). h kept in LDS as f32
// (state) + f16 (WMMA operand). Recurrent gh = h @ Whh^T via
// v_wmma_f32_16x16x32_f16: K=32/instr -> 8 WMMAs per 16x16 tile per step.
// ---------------------------------------------------------------------------
struct GruArgs {
  const float* gi[4];    // precomputed x@Wih^T+bih, [S,B,768]
  const f16_t* whh[4];   // [768,256] f16
  const float* bhh[4];   // [768]
  bf16_t*      hid[4];   // output base (+256 col offset for backward)
  int          bwd[4];
};

__global__ __launch_bounds__(512)
void gru_scan(GruArgs args) {
  const int blk = blockIdx.x;
  const float* gi  = args.gi[blk];
  const f16_t* whh = args.whh[blk];
  const float* bhh = args.bhh[blk];
  bf16_t* hid = args.hid[blk];
  const int bwd = args.bwd[blk];

  extern __shared__ float sm[];
  float* h   = sm;                         // [32][256] f32 state
  float* gh  = sm + 32 * 256;              // [32][768]
  f16_t* h16 = (f16_t*)(sm + 32 * 256 + 32 * 768);  // [32][256] f16 shadow

  const int tid = threadIdx.x;
  const int lane = tid & 31, wv = tid >> 5;  // 16 waves
  const int g = lane >> 4, c = lane & 15;

  for (int i = tid; i < 32 * 256; i += 512) { h[i] = 0.f; h16[i] = (f16_t)0.f; }
  __syncthreads();

  for (int step = 0; step < S_; ++step) {
    const int s = bwd ? (S_ - 1 - step) : step;
    // --- gh = h @ Whh^T + bhh : 96 tiles of 16x16, 6 per wave ---
    #pragma unroll
    for (int i = 0; i < 6; ++i) {
      const int tl = wv + 16 * i;
      const int m0 = (tl & 1) << 4;
      const int n0 = (tl >> 1) << 4;
      Acc8 acc;
      const float bb = bhh[n0 + c];
      #pragma unroll
      for (int j = 0; j < 8; ++j) acc.f[j] = bb;
      const f16_t* hrow = h16 + (m0 + c) * 256;
      const f16_t* wrow = whh + (size_t)(n0 + c) * 256;
      #pragma unroll
      for (int kk = 0; kk < 256; kk += 32) {
        FragH ah, wh;
        ah.q[0] = *(const uint4*)(hrow + kk + g * 8);
        ah.q[1] = *(const uint4*)(hrow + kk + 16 + g * 8);
        wh.q[0] = *(const uint4*)(wrow + kk + g * 8);
        wh.q[1] = *(const uint4*)(wrow + kk + 16 + g * 8);
        acc.v = __builtin_amdgcn_wmma_f32_16x16x32_f16(false, ah.v, false, wh.v,
                                                       (short)0, acc.v, false, false);
      }
      #pragma unroll
      for (int j = 0; j < 8; ++j)
        gh[(m0 + g * 8 + j) * 768 + n0 + c] = acc.f[j];
    }
    __syncthreads();
    // --- gates + state update ---
    const float* gir = gi + (size_t)s * 32 * 768;
    for (int idx = tid; idx < 32 * 256; idx += 512) {
      const int b = idx >> 8, j = idx & 255;
      const float* gb = gir + b * 768;
      const float r = sigmoidf_(gb[j]       + gh[b * 768 + j]);
      const float z = sigmoidf_(gb[256 + j] + gh[b * 768 + 256 + j]);
      const float n = tanhf(gb[512 + j] + r * gh[b * 768 + 512 + j]);
      const float hn = (1.f - z) * n + z * h[b * 256 + j];
      h[b * 256 + j] = hn;
      h16[b * 256 + j] = (f16_t)hn;
      hid[((size_t)s * 32 + b) * 512 + j] = (bf16_t)hn;
    }
    __syncthreads();
  }
}

// ---------------------------------------------------------------------------
// Tape scan: one block per batch element, tape slice (64KB) LDS-resident.
// Per-step q-proj/value vectors streamed into LDS with async-to-LDS copies.
// ---------------------------------------------------------------------------
__global__ __launch_bounds__(256)
void tape_scan(const float* __restrict__ val, const float* __restrict__ qp,
               const float* __restrict__ actp, float* __restrict__ tape_g,
               bf16_t* __restrict__ outs, int store_tape) {
  const int b = blockIdx.x, tid = threadIdx.x;
  extern __shared__ float sm[];
  float* tp  = sm;                 // [32][512] tape[l][t*128+c]
  float* qld = tp + 32 * 512;      // [512]
  float* vld = qld + 512;          // [512]
  float* rp  = vld + 512;          // [128] rpos[l*4+t]
  float* wp  = rp + 128;
  float* rn  = wp + 128;
  float* wn  = rn + 128;
  float* jp  = wn + 128;
  float* sc  = jp + 128;
  float* ald = sc + 128;           // [40]

  for (int i = tid; i < L_ * D_; i += 256) {
    const int l = i >> 9, rest = i & 511;
    tp[i] = tape_g[((size_t)l * 32 + b) * 512 + rest];
  }
  for (int p = tid; p < 128; p += 256) {
    const float v0 = ((p >> 2) == 0) ? 1.f : 0.f;
    rp[p] = v0; wp[p] = v0;
  }
  __syncthreads();

  for (int s = 0; s < S_; ++s) {
    const size_t rbase = (size_t)s * 32 + b;
    // stream q-proj + values for this step into LDS (one b128 per thread)
    if (tid < 128) async_load_b128(lds_off(&qld[tid * 4]), qp + rbase * 512 + tid * 4);
    else           async_load_b128(lds_off(&vld[(tid - 128) * 4]), val + rbase * 512 + (tid - 128) * 4);
    if (tid < 40)  ald[tid] = actp[rbase * 40 + tid];
    wait_async0();
    __syncthreads();
    // write head + read out (each (t,c) column owned by one thread)
    for (int o = tid; o < 512; o += 256) {
      const int t = o >> 7;
      const float rw1 = ald[t * 10 + 9];
      float ov = 0.f;
      for (int l = 0; l < 32; ++l) ov += tp[l * 512 + o] * wp[l * 4 + t];
      const float dv = (vld[o] - ov) * rw1;
      float ro = 0.f;
      for (int l = 0; l < 32; ++l) {
        const float nv = tp[l * 512 + o] + wp[l * 4 + t] * dv;
        tp[l * 512 + o] = nv;
        ro += nv * rp[l * 4 + t];
      }
      if (outs) outs[rbase * 512 + o] = (bf16_t)(ro * ald[t * 10 + 8]);
    }
    __syncthreads();
    // jump scores: score[l,t] = tape[l,t,:] . qproj[t,:]
    if (tid < 128) {
      const int l = tid >> 2, t = tid & 3;
      const float* trow = tp + l * 512 + t * 128;
      const float* qrow = qld + t * 128;
      float acc = 0.f;
      for (int cc = 0; cc < 128; ++cc) acc += trow[cc] * qrow[cc];
      sc[tid] = acc;
    }
    __syncthreads();
    if (tid < 4) {
      const int t = tid;
      float mx = -1e30f;
      for (int l = 0; l < 32; ++l) mx = fmaxf(mx, sc[l * 4 + t]);
      float sum = 0.f;
      for (int l = 0; l < 32; ++l) { float e = __expf(sc[l * 4 + t] - mx); jp[l * 4 + t] = e; sum += e; }
      const float inv = 1.f / sum;
      for (int l = 0; l < 32; ++l) jp[l * 4 + t] *= inv;
    }
    __syncthreads();
    if (tid < 128) {
      const int l = tid >> 2, t = tid & 3;
      const float prv = rp[((l + 1) & 31) * 4 + t];   // roll(rpos,-1)
      const float nxt = rp[((l - 1) & 31) * 4 + t];   // roll(rpos,+1)
      const float cj = jp[tid];
      wn[tid] = prv * ald[t * 10 + 4] + wp[tid] * ald[t * 10 + 5]
              + nxt * ald[t * 10 + 6] + cj * ald[t * 10 + 7];
      rn[tid] = prv * ald[t * 10 + 0] + rp[tid] * ald[t * 10 + 1]
              + nxt * ald[t * 10 + 2] + cj * ald[t * 10 + 3];
    }
    __syncthreads();
    if (tid < 128) { rp[tid] = rn[tid]; wp[tid] = wn[tid]; }
    __syncthreads();
  }
  if (store_tape) {
    for (int i = tid; i < L_ * D_; i += 256) {
      const int l = i >> 9, rest = i & 511;
      tape_g[((size_t)l * 32 + b) * 512 + rest] = tp[i];
    }
  }
}

// ---------------------------------------------------------------------------
extern "C" void kernel_launch(void* const* d_in, const int* in_sizes, int n_in,
                              void* d_out, int out_size, void* d_ws, size_t ws_size,
                              hipStream_t stream) {
  (void)in_sizes; (void)n_in; (void)out_size; (void)ws_size;
  const int*   inp  = (const int*)d_in[0];
  const float* emb  = (const float*)d_in[1];
  const float* Wemb = (const float*)d_in[2];
  const float* bemb = (const float*)d_in[3];
  const float* g1   = (const float*)d_in[4];
  const float* b1   = (const float*)d_in[5];
  const float* Wih[4] = {(const float*)d_in[6],  (const float*)d_in[10],
                         (const float*)d_in[24], (const float*)d_in[28]};
  const float* Whh[4] = {(const float*)d_in[7],  (const float*)d_in[11],
                         (const float*)d_in[25], (const float*)d_in[29]};
  const float* bih[4] = {(const float*)d_in[8],  (const float*)d_in[12],
                         (const float*)d_in[26], (const float*)d_in[30]};
  const float* bhh[4] = {(const float*)d_in[9],  (const float*)d_in[13],
                         (const float*)d_in[27], (const float*)d_in[31]};
  const float* Wa[2] = {(const float*)d_in[14], (const float*)d_in[32]};
  const float* ba[2] = {(const float*)d_in[15], (const float*)d_in[33]};
  const float* Wv[2] = {(const float*)d_in[16], (const float*)d_in[34]};
  const float* bv[2] = {(const float*)d_in[17], (const float*)d_in[35]};
  const float* Wo2 = (const float*)d_in[36];
  const float* bo2 = (const float*)d_in[37];
  const float* Wk[2] = {(const float*)d_in[20], (const float*)d_in[38]};
  const float* Wq[2] = {(const float*)d_in[22], (const float*)d_in[40]};
  const float* bq[2] = {(const float*)d_in[23], (const float*)d_in[41]};
  const float* g2  = (const float*)d_in[42];
  const float* b2v = (const float*)d_in[43];
  const float* Wfc = (const float*)d_in[44];
  const float* bfc = (const float*)d_in[45];

  // ---- workspace layout ----
  char* wsB = (char*)d_ws;
  size_t off = 0;
  auto take = [&](size_t bytes) -> void* {
    void* p = wsB + off;
    off += (bytes + 255) & ~(size_t)255;
    return p;
  };
  const size_t RC = (size_t)S_ * B_;   // 8192 rows
  bf16_t* wemb_h = (bf16_t*)take((size_t)D_ * D_ * 2);
  bf16_t* wih_h[4]; for (int i = 0; i < 4; ++i) wih_h[i] = (bf16_t*)take((size_t)G3_ * D_ * 2);
  f16_t*  whh_h[4]; for (int i = 0; i < 4; ++i) whh_h[i] = (f16_t*) take((size_t)G3_ * H_ * 2);
  bf16_t* wa_h[2];  for (int i = 0; i < 2; ++i) wa_h[i]  = (bf16_t*)take((size_t)48 * D_ * 2);
  bf16_t* wv_h[2];  for (int i = 0; i < 2; ++i) wv_h[i]  = (bf16_t*)take((size_t)D_ * D_ * 2);
  bf16_t* wq_h[2];  for (int i = 0; i < 2; ++i) wq_h[i]  = (bf16_t*)take((size_t)D_ * D_ * 2);
  bf16_t* wkT_h[2]; for (int i = 0; i < 2; ++i) wkT_h[i] = (bf16_t*)take((size_t)HD_ * HD_ * 2);
  bf16_t* wo2_h = (bf16_t*)take((size_t)D_ * D_ * 2);
  bf16_t* wfc_h = (bf16_t*)take((size_t)V_ * D_ * 2);
  float* bapad[2]; for (int i = 0; i < 2; ++i) bapad[i] = (float*)take(48 * 4);
  bf16_t* src_h = (bf16_t*)take(RC * D_ * 2);
  bf16_t* hid_h[2]; for (int i = 0; i < 2; ++i) hid_h[i] = (bf16_t*)take(RC * D_ * 2);
  float* act[2];  for (int i = 0; i < 2; ++i) act[i]  = (float*)take(RC * 48 * 4);
  float* actp[2]; for (int i = 0; i < 2; ++i) actp[i] = (float*)take(RC * 40 * 4);
  float* tape = (float*)take((size_t)L_ * B_ * D_ * 4);
  bf16_t* outs_h = (bf16_t*)take(RC * D_ * 2);
  // reused big region (96 MB)
  char* R = (char*)take((size_t)4 * RC * G3_ * 4);
  bf16_t* rowsA  = (bf16_t*)R;                               // phase 1
  float*  srcpre = (float*)(R + RC * D_ * 2);
  float*  gi[4];
  for (int i = 0; i < 4; ++i) gi[i] = (float*)(R + (size_t)i * RC * G3_ * 4);   // phase 2
  bf16_t* qh[2]  = {(bf16_t*)(R + 0),                 (bf16_t*)(R + (size_t)8  * 1024 * 1024)};
  float*  valb[2]= {(float*) (R + (size_t)16 * 1024 * 1024), (float*)(R + (size_t)32 * 1024 * 1024)};
  float*  qpb[2] = {(float*) (R + (size_t)48 * 1024 * 1024), (float*)(R + (size_t)64 * 1024 * 1024)};
  float*  out2f  = (float*) (R + 0);                          // phase 4 (qh dead)
  bf16_t* hfin_h = (bf16_t*)(R + (size_t)16 * 1024 * 1024);   // phase 4 (val dead)

  auto cvt = [&](const float* x, bf16_t* y, size_t n) {
    cvt_bf16<<<dim3((unsigned)((n + 255) / 256)), 256, 0, stream>>>(x, y, (int)n);
  };

  // ---- weight conversion ----
  cvt(Wemb, wemb_h, (size_t)D_ * D_);
  for (int i = 0; i < 4; ++i) cvt(Wih[i], wih_h[i], (size_t)G3_ * D_);
  for (int i = 0; i < 4; ++i)
    cvt_f16<<<dim3((unsigned)((G3_ * H_ + 255) / 256)), 256, 0, stream>>>(
        Whh[i], whh_h[i], G3_ * H_);
  for (int i = 0; i < 2; ++i) cvt(Wv[i], wv_h[i], (size_t)D_ * D_);
  for (int i = 0; i < 2; ++i) cvt(Wq[i], wq_h[i], (size_t)D_ * D_);
  cvt(Wo2, wo2_h, (size_t)D_ * D_);
  cvt(Wfc, wfc_h, (size_t)V_ * D_);
  for (int i = 0; i < 2; ++i)
    pad_wa<<<96, 256, 0, stream>>>(Wa[i], ba[i], wa_h[i], bapad[i]);
  for (int i = 0; i < 2; ++i)
    cvt_transpose128<<<64, 256, 0, stream>>>(Wk[i], wkT_h[i]);
  hipMemsetAsync(tape, 0, (size_t)L_ * B_ * D_ * 4, stream);

  // ---- embed -> proj -> LN ----
  gather_emb<<<(unsigned)RC, 256, 0, stream>>>(inp, emb, rowsA);
  gemm_bf16<0><<<dim3(D_ / 128, (unsigned)(RC / 16)), 256, 0, stream>>>(
      rowsA, wemb_h, bemb, srcpre, (int)RC, D_, D_);
  ln_kernel<<<(unsigned)RC, 256, 0, stream>>>(srcpre, g1, b1, src_h, 0);

  // ---- gi = src @ Wih^T + bih (x4) ----
  for (int i = 0; i < 4; ++i)
    gemm_bf16<0><<<dim3(G3_ / 128, (unsigned)(RC / 16)), 256, 0, stream>>>(
        src_h, wih_h[i], bih[i], gi[i], (int)RC, G3_, D_);

  // ---- persistent GRU scans (4 concurrent: tm x dir) ----
  GruArgs ga;
  for (int i = 0; i < 4; ++i) { ga.gi[i] = gi[i]; ga.whh[i] = whh_h[i]; ga.bhh[i] = bhh[i]; }
  ga.hid[0] = hid_h[0];       ga.bwd[0] = 0;
  ga.hid[1] = hid_h[0] + 256; ga.bwd[1] = 1;
  ga.hid[2] = hid_h[1];       ga.bwd[2] = 0;
  ga.hid[3] = hid_h[1] + 256; ga.bwd[3] = 1;
  gru_scan<<<4, 512, 147456, stream>>>(ga);   // h(32K) + gh(96K) + h16(16K)

  // ---- per-module projections ----
  for (int i = 0; i < 2; ++i) {
    gemm_bf16<1><<<dim3(D_ / 128, (unsigned)(RC / 16)), 256, 0, stream>>>(
        hid_h[i], wq_h[i], bq[i], qh[i], (int)RC, D_, D_);
    gemm_bf16<0><<<dim3(D_ / 128, (unsigned)(RC / 16)), 256, 0, stream>>>(
        src_h, wv_h[i], bv[i], valb[i], (int)RC, D_, D_);
    gemm_bf16<0><<<dim3(1, (unsigned)(RC / 16)), 256, 0, stream>>>(
        hid_h[i], wa_h[i], bapad[i], act[i], (int)RC, 48, D_);
    act_proc<<<128, 256, 0, stream>>>(act[i], actp[i]);
    // qproj = q @ Wk  (rows = (s,b,t), K = N = 128)
    gemm_bf16<0><<<dim3(1, (unsigned)(RC * T_ / 16)), 256, 0, stream>>>(
        qh[i], wkT_h[i], nullptr, qpb[i], (int)(RC * T_), HD_, HD_);
  }

  // ---- tape scans (TM1 tape feeds TM2; TM1 output is dead) ----
  const unsigned tapeLds = (32 * 512 + 512 + 512 + 128 * 6 + 40) * 4;
  tape_scan<<<B_, 256, tapeLds, stream>>>(valb[0], qpb[0], actp[0], tape, nullptr, 1);
  tape_scan<<<B_, 256, tapeLds, stream>>>(valb[1], qpb[1], actp[1], tape, outs_h, 0);

  // ---- out = outs @ Wo2^T + bo2 -> LN+ReLU -> logits (transposed store) ----
  gemm_bf16<0><<<dim3(D_ / 128, (unsigned)(RC / 16)), 256, 0, stream>>>(
      outs_h, wo2_h, bo2, out2f, (int)RC, D_, D_);
  ln_kernel<<<(unsigned)RC, 256, 0, stream>>>(out2f, g2, b2v, hfin_h, 1);
  gemm_bf16<2><<<dim3(V_ / 128, (unsigned)(RC / 16)), 256, 0, stream>>>(
      hfin_h, wfc_h, bfc, d_out, (int)RC, V_, D_);
}
